// MoELoraModel_32006096290495
// MI455X (gfx1250) — matrix operand order repaired and checked by reading, not compiled, not run on hardware
//
#include <hip/hip_runtime.h>
#include <math.h>

typedef __attribute__((ext_vector_type(16))) _Float16 v16h;
typedef __attribute__((ext_vector_type(8)))  float    v8f;

#define B_TOK 16384
#define SEQ   64
#define H_DIM 256
#define N_EXP 8

// CDNA5 hardware tanh (V_TANH_F32, single TRANS op) when the toolchain has it.
#if defined(__has_builtin)
#if __has_builtin(__builtin_amdgcn_tanhf)
#define FAST_TANH(x) __builtin_amdgcn_tanhf(x)
#endif
#endif
#ifndef FAST_TANH
#define FAST_TANH(x) tanhf(x)
#endif

// ---------------------------------------------------------------------------
// Kernel 1: router — logits = xr @ W_router, top-2 softmax -> dense gate [B,8]
// One wave32 per token; lane-strided MAC + shfl_xor reduction.
// ---------------------------------------------------------------------------
__global__ __launch_bounds__(256) void router_kernel(const float* __restrict__ xr,
                                                     const float* __restrict__ Wr,
                                                     float* __restrict__ gate) {
    int wave = threadIdx.x >> 5;
    int lane = threadIdx.x & 31;
    int b = blockIdx.x * 8 + wave;

    float acc[N_EXP];
#pragma unroll
    for (int e = 0; e < N_EXP; ++e) acc[e] = 0.0f;

    for (int k = lane; k < H_DIM; k += 32) {
        float xv = xr[(size_t)b * H_DIM + k];
#pragma unroll
        for (int e = 0; e < N_EXP; ++e) acc[e] += xv * Wr[k * N_EXP + e];
    }
#pragma unroll
    for (int e = 0; e < N_EXP; ++e) {
#pragma unroll
        for (int off = 16; off > 0; off >>= 1)
            acc[e] += __shfl_xor(acc[e], off, 32);
    }
    if (lane == 0) {
        float best = -INFINITY, sec = -INFINITY;
        int bi = 0, si = 1;
        for (int e = 0; e < N_EXP; ++e) {
            float v = acc[e];
            if (v > best)      { sec = best; si = bi; best = v; bi = e; }
            else if (v > sec)  { sec = v; si = e; }
        }
        // softmax over (best, sec); best >= sec so exp() never overflows
        float e2  = __expf(sec - best);
        float inv = 1.0f / (1.0f + e2);
        float g[N_EXP];
        for (int e = 0; e < N_EXP; ++e) g[e] = 0.0f;
        g[bi] = inv;
        g[si] = e2 * inv;
        for (int e = 0; e < N_EXP; ++e) gate[(size_t)b * N_EXP + e] = g[e];
    }
}

// ---------------------------------------------------------------------------
// Kernel 2: repack W_experts f32 [E][K=256][N=256] into f16 B-fragment layout:
// Wp[e][kt(8)][nt(16)][lane(32)][i(16)], per CDNA5 ISA the 32x16 16-bit
// B tile: lanes 0-15 -> K 0..7 / 16..23 of column N=lane,
//         lanes 16-31 -> K 8..15 / 24..31 of column N=lane-16.
// Each lane's fragment becomes one contiguous 32B chunk.
// ---------------------------------------------------------------------------
__global__ __launch_bounds__(32) void wpack_kernel(const float* __restrict__ W,
                                                   _Float16* __restrict__ Wp) {
    int lane = threadIdx.x;
    int bi = blockIdx.x;            // e*128 + kt*16 + nt
    int e  = bi >> 7;
    int kt = (bi >> 4) & 7;
    int nt = bi & 15;
    int n     = nt * 16 + (lane & 15);
    int kbase = kt * 32 + ((lane & 16) ? 8 : 0);
    v16h v;
#pragma unroll
    for (int i = 0; i < 16; ++i)
        v[i] = (_Float16)W[((size_t)e * H_DIM + (kbase + ((i & 7) | ((i & 8) << 1)))) * H_DIM + n];
    *(v16h*)(Wp + ((size_t)bi * 32 + lane) * 16) = v;
}

// ---------------------------------------------------------------------------
// Kernel 3: pooled = mean(x, axis=1), convert to f16, pack A-fragment layout:
// Ap[bt(B/16)][kt(8)][lane(32)][i(16)]; per ISA table the 16x32 16-bit A tile:
// lane 0-15 row M=lane: i=0..7 -> K 0..7, i=8..15 -> K 16..23;
// lane 16-31 row M=lane-16: K 8..15 / 24..31.
// The 1 GiB x stream uses non-temporal loads (read-once -> don't pollute L2).
// ---------------------------------------------------------------------------
__global__ __launch_bounds__(256) void pool_pack_kernel(const float* __restrict__ x,
                                                        _Float16* __restrict__ Ap) {
    __shared__ _Float16 p[16 * H_DIM];   // 8 KB
    int t  = threadIdx.x;                // h index
    int bt = blockIdx.x;                 // 16-token tile
    const float scale = 1.0f / (float)SEQ;

    for (int m = 0; m < 16; ++m) {
        const float* src = x + ((size_t)(bt * 16 + m) * SEQ) * H_DIM + t;
        float sum = 0.0f;
#pragma unroll 8
        for (int s = 0; s < SEQ; ++s)
            sum += __builtin_nontemporal_load(src + (size_t)s * H_DIM);
        p[m * H_DIM + t] = (_Float16)(sum * scale);
    }
    __syncthreads();

    int kt    = t >> 5;                  // wave id == ktile
    int lane  = t & 31;
    int M     = lane & 15;
    int khalf = (lane & 16) ? 8 : 0;
    v16h a;
#pragma unroll
    for (int i = 0; i < 16; ++i) {
        int K = kt * 32 + khalf + ((i & 7) | ((i & 8) << 1)); // 0..7 -> K, 8..15 -> K+16
        a[i] = p[M * H_DIM + K];
    }
    *(v16h*)(Ap + (((size_t)bt * 8 + kt) * 32 + lane) * 16) = a;
}

// ---------------------------------------------------------------------------
// Kernel 4: dense 8-expert WMMA GEMM + bias + hardware tanh + gated combine.
// 32 tokens / block, 8 waves; wave owns (mtile, 4 ntiles).
// 256 x v_wmma_f32_16x16x32_f16 per wave; A fragments register-resident;
// gate row-values hoisted into registers per expert (no LDS reads in nt loop).
// ---------------------------------------------------------------------------
__global__ __launch_bounds__(256) void moe_wmma_kernel(const _Float16* __restrict__ Ap,
                                                       const _Float16* __restrict__ Wp,
                                                       const float* __restrict__ gate,
                                                       const float* __restrict__ bexp,
                                                       float* __restrict__ out) {
    __shared__ float gate_s[32 * N_EXP];     // 1 KB
    __shared__ float bias_s[N_EXP * H_DIM];  // 8 KB
    int t  = threadIdx.x;
    int b0 = blockIdx.x * 32;

    gate_s[t] = gate[(size_t)b0 * N_EXP + t];           // 32 tokens * 8 = 256
    for (int i = t; i < N_EXP * H_DIM; i += 256) bias_s[i] = bexp[i];
    __syncthreads();

    int wave = t >> 5, lane = t & 31;
    int mt   = wave & 1;                // which 16-token subtile
    int ntb  = (wave >> 1) * 4;         // first of 4 owned ntiles
    int bt   = blockIdx.x * 2 + mt;
    int mrow = mt * 16 + ((lane & 16) >> 1);   // C-frag row base for this lane

    v16h aF[8];
#pragma unroll
    for (int kt = 0; kt < 8; ++kt)
        aF[kt] = *(const v16h*)(Ap + (((size_t)bt * 8 + kt) * 32 + lane) * 16);

    float accP[4][8];
#pragma unroll
    for (int nt = 0; nt < 4; ++nt)
#pragma unroll
        for (int r = 0; r < 8; ++r) accP[nt][r] = 0.0f;

    for (int e = 0; e < N_EXP; ++e) {
        // per-row gate weights for this expert, hoisted to registers
        float g[8];
#pragma unroll
        for (int r = 0; r < 8; ++r)
            g[r] = gate_s[(mrow + r) * N_EXP + e];

#pragma unroll
        for (int nt = 0; nt < 4; ++nt) {
            v8f c = {};
#pragma unroll
            for (int kt = 0; kt < 8; ++kt) {
                v16h bF = *(const v16h*)(Wp +
                    ((((size_t)e * 8 + kt) * 16 + (ntb + nt)) * 32 + lane) * 16);
                c = __builtin_amdgcn_wmma_f32_16x16x32_f16(
                        false, aF[kt], false, bF, (short)0, c, false, false);
            }
            float bias = bias_s[e * H_DIM + (ntb + nt) * 16 + (lane & 15)];
#pragma unroll
            for (int r = 0; r < 8; ++r)
                accP[nt][r] += g[r] * FAST_TANH(c[r] + bias);
        }
    }
#pragma unroll
    for (int nt = 0; nt < 4; ++nt) {
        int n = (ntb + nt) * 16 + (lane & 15);
#pragma unroll
        for (int r = 0; r < 8; ++r)
            out[(size_t)(b0 + mrow + r) * H_DIM + n] = accP[nt][r];
    }
}

// ---------------------------------------------------------------------------
extern "C" void kernel_launch(void* const* d_in, const int* in_sizes, int n_in,
                              void* d_out, int out_size, void* d_ws, size_t ws_size,
                              hipStream_t stream) {
    const float* x  = (const float*)d_in[0];   // [B,S,H]
    const float* xr = (const float*)d_in[1];   // [B,H]
    const float* Wr = (const float*)d_in[2];   // [H,E]
    const float* We = (const float*)d_in[3];   // [E,H,H]
    const float* be = (const float*)d_in[4];   // [E,H]
    float* out = (float*)d_out;                // [B,H]

    char* ws = (char*)d_ws;
    _Float16* Ap   = (_Float16*)ws;                                   // 8 MB
    _Float16* Wp   = (_Float16*)(ws + (size_t)8 * 1024 * 1024);       // 1 MB
    float*    gate = (float*)   (ws + (size_t)9 * 1024 * 1024);       // 512 KB

    wpack_kernel    <<<N_EXP * 8 * 16, 32,  0, stream>>>(We, Wp);
    router_kernel   <<<B_TOK / 8,      256, 0, stream>>>(xr, Wr, gate);
    pool_pack_kernel<<<B_TOK / 16,     256, 0, stream>>>(x, Ap);
    moe_wmma_kernel <<<B_TOK / 32,     256, 0, stream>>>(Ap, Wp, gate, be, out);
}
